// ProteinTransformerDenoiserModelV1_721554505891
// MI455X (gfx1250) — matrix operand order repaired and checked by reading, not compiled, not run on hardware
//
#include <hip/hip_runtime.h>
#include <hip/hip_bf16.h>
#include <math.h>

// ---------------------------------------------------------------------------
// Types for CDNA5 WMMA (wave32): D(16x16 f32) = A(16x32 bf16) x B(32x16 bf16) + C
// ---------------------------------------------------------------------------
typedef __attribute__((ext_vector_type(16))) __bf16 v16bf;
typedef __attribute__((ext_vector_type(8)))  __bf16 v8bf;
typedef __attribute__((ext_vector_type(8)))  float  v8f;

#define BB    2
#define LL    2048
#define DD    512
#define HH    8
#define EE    64
#define DFF   1024
#define NLAY  4
#define ROWS  (BB*LL)          // 4096
#define EPSF  1e-6f

static __device__ inline v8f wmma_bf16(v16bf a, v16bf b, v8f c) {
  // v_wmma_f32_16x16x32_bf16  (neg_a, A, neg_b, B, c_mod, C, reuse_a, reuse_b)
  return __builtin_amdgcn_wmma_f32_16x16x32_bf16(false, a, false, b, (short)0, c,
                                                 false, false);
}

// A-operand fragment: lane (m = lane&15, khalf = lane>>4) holds K =
// {base..base+15} with base = kk + khalf*16.  `p` points at
// row_ptr + kk + khalf*16.  Elements 0..15 -> K base..base+15
static __device__ inline v16bf load_a_frag(const __bf16* p) {
  v8bf lo = *(const v8bf*)(p);
  v8bf hi = *(const v8bf*)(p + 8);
  v16bf r;
#pragma unroll
  for (int i = 0; i < 8; ++i) { r[i] = lo[i]; r[i + 8] = hi[i]; }
  return r;
}

// ---------------------------------------------------------------------------
// 1) cond gains: g[layer][b][d] = cond[b] . wn[layer][:,d] + 1
// ---------------------------------------------------------------------------
__global__ void cond_gain_kernel(const float* __restrict__ cond,
                                 const float* __restrict__ wn,
                                 float* __restrict__ g) {
  int idx = blockIdx.x * blockDim.x + threadIdx.x;     // NLAY*BB*DD = 4096
  if (idx >= NLAY * BB * DD) return;
  int d = idx % DD, b = (idx / DD) % BB, layer = idx / (BB * DD);
  const float* w = wn + (size_t)layer * DD * DD;
  const float* c = cond + (size_t)b * DD;
  float acc = 1.0f;
#pragma unroll 4
  for (int i = 0; i < DD; ++i) acc += c[i] * w[(size_t)i * DD + d];
  g[idx] = acc;
}

// ---------------------------------------------------------------------------
// 2) weight convert: W[K][N] f32 -> Wt[N][K] bf16 (so B-operand is contiguous)
// ---------------------------------------------------------------------------
__global__ void w_to_bf16_t_kernel(const float* __restrict__ W,
                                   __bf16* __restrict__ Wt, int K, int N) {
  int idx = blockIdx.x * blockDim.x + threadIdx.x;
  if (idx >= K * N) return;
  int k = idx % K, n = idx / K;
  Wt[(size_t)n * K + k] = (__bf16)W[(size_t)k * N + n];
}

// ---------------------------------------------------------------------------
// 3) RMSNorm (+gain) over D=512, one wave per row, 16 elems/lane, bf16|f32 out
// ---------------------------------------------------------------------------
template <typename OUT, bool PER_BATCH>
__global__ void rmsnorm_kernel(const float* __restrict__ x,
                               const float* __restrict__ gain,
                               OUT* __restrict__ y) {
  int row  = blockIdx.x * 8 + (threadIdx.x >> 5);
  int lane = threadIdx.x & 31;
  const float* xr = x + (size_t)row * DD;
  float v[16]; float ss = 0.f;
#pragma unroll
  for (int i = 0; i < 16; ++i) { v[i] = xr[i * 32 + lane]; ss += v[i] * v[i]; }
#pragma unroll
  for (int off = 1; off < 32; off <<= 1) ss += __shfl_xor(ss, off);
  float inv = rsqrtf(ss * (1.0f / DD) + EPSF);
  const float* g = PER_BATCH ? (gain + (size_t)(row / LL) * DD) : gain;
  OUT* yr = y + (size_t)row * DD;
#pragma unroll
  for (int i = 0; i < 16; ++i)
    yr[i * 32 + lane] = (OUT)(v[i] * g[i * 32 + lane] * inv);
}

// ---------------------------------------------------------------------------
// 4) GEMM: C[M,N] f32 = A[M,K] bf16 @ Wt[N,K] bf16 (+ optional residual)
//    One wave -> 32(M) x 64(N) strip: 8 accumulator tiles, each B-fragment
//    feeds two WMMAs (B reuse across the two 16-row halves).  Weights stay
//    L2-resident (192 MB), activations are streamed with prefetch.
// ---------------------------------------------------------------------------
__global__ void gemm_bf16_wmma_kernel(const __bf16* __restrict__ A,
                                      const __bf16* __restrict__ Wt,
                                      const float* __restrict__ resid,
                                      float* __restrict__ C,
                                      int M, int N, int K) {
  int wv = threadIdx.x >> 5, lane = threadIdx.x & 31;
  int gw = blockIdx.x * 8 + wv;
  int strips = N >> 6;
  int tm = gw / strips, tn = gw % strips;
  if (tm * 32 >= M) return;                       // wave-uniform guard
  int ml = lane & 15, kh = lane >> 4;

  const __bf16* arow0 = A + (size_t)(tm * 32 + ml) * K;
  const __bf16* arow1 = arow0 + (size_t)16 * K;
  v8f acc0[4] = {}, acc1[4] = {};
  for (int kk = 0; kk < K; kk += 32) {
    if (kk + 32 < K) {                            // stream-ahead (global_prefetch_b8)
      __builtin_prefetch((const void*)(arow0 + kk + 32), 0, 3);
      __builtin_prefetch((const void*)(arow1 + kk + 32), 0, 3);
    }
    v16bf a0 = load_a_frag(arow0 + kk + kh * 16);
    v16bf a1 = load_a_frag(arow1 + kk + kh * 16);
#pragma unroll
    for (int nt = 0; nt < 4; ++nt) {
      const __bf16* wp =
          Wt + (size_t)(tn * 64 + nt * 16 + ml) * K + kk + kh * 16;
      v16bf b = *(const v16bf*)wp;                // 16 contiguous K, 32B aligned
      acc0[nt] = wmma_bf16(a0, b, acc0[nt]);
      acc1[nt] = wmma_bf16(a1, b, acc1[nt]);
    }
  }
#pragma unroll
  for (int nt = 0; nt < 4; ++nt)
#pragma unroll
    for (int r = 0; r < 8; ++r) {
      int n = tn * 64 + nt * 16 + ml;
      int m0 = tm * 32 + r + (kh ? 8 : 0);
      int m1 = m0 + 16;
      float v0 = acc0[nt][r], v1 = acc1[nt][r];
      if (resid) {
        v0 += resid[(size_t)m0 * N + n];
        v1 += resid[(size_t)m1 * N + n];
      }
      C[(size_t)m0 * N + n] = v0;
      C[(size_t)m1 * N + n] = v1;
    }
}

// ---------------------------------------------------------------------------
// 5) q/k head RMSNorm + RoPE (+1/sqrt(E) into q), v transpose. 1 wave/(b,h,l)
//    lane holds e = lane and e = lane+32 (exactly the RoPE pairing).
// ---------------------------------------------------------------------------
__global__ void qk_prep_kernel(const float* __restrict__ qkv,
                               const float* __restrict__ qs,   // [H], this layer
                               __bf16* __restrict__ qb, __bf16* __restrict__ kb,
                               __bf16* __restrict__ vt) {
  int gw = blockIdx.x * 8 + (threadIdx.x >> 5);   // over B*H*L
  int lane = threadIdx.x & 31;
  int l = gw % LL, h = (gw / LL) % HH, b = gw / (LL * HH);
  const float* base = qkv + (size_t)(b * LL + l) * (3 * DD) + h * EE;
  float q0 = base[lane],        q1 = base[lane + 32];
  float k0 = base[DD + lane],   k1 = base[DD + lane + 32];
  float v0 = base[2*DD + lane], v1 = base[2*DD + lane + 32];

  float sq = q0 * q0 + q1 * q1, sk = k0 * k0 + k1 * k1;
#pragma unroll
  for (int off = 1; off < 32; off <<= 1) {
    sq += __shfl_xor(sq, off);
    sk += __shfl_xor(sk, off);
  }
  // hs = exp(0.5*min(qs, log(100)) - 0.25*ln(64))
  float hs = __expf(0.5f * fminf(qs[h], 4.6051702f) - 0.25f * 4.1588831f);
  float qi = hs * rsqrtf(sq * (1.0f / EE) + EPSF);
  float ki = hs * rsqrtf(sk * (1.0f / EE) + EPSF);
  float qn0 = q0 * qi, qn1 = q1 * qi, kn0 = k0 * ki, kn1 = k1 * ki;

  // RoPE: inv = 10000^(-2*lane/64), freq = l * inv
  float inv = __expf(-((float)(2 * lane) / (float)EE) * 9.2103404f); // ln 1e4
  float fr = (float)l * inv, c = __cosf(fr), s = __sinf(fr);
  float qo0 = (qn0 * c - qn1 * s) * 0.125f;       // fold 1/sqrt(64)
  float qo1 = (qn0 * s + qn1 * c) * 0.125f;
  float ko0 = kn0 * c - kn1 * s;
  float ko1 = kn0 * s + kn1 * c;

  size_t ro = ((size_t)(b * HH + h) * LL + l) * EE;
  qb[ro + lane] = (__bf16)qo0; qb[ro + lane + 32] = (__bf16)qo1;
  kb[ro + lane] = (__bf16)ko0; kb[ro + lane + 32] = (__bf16)ko1;
  size_t vo = (size_t)(b * HH + h) * EE;
  vt[(vo + lane)      * LL + l] = (__bf16)v0;     // vT[b][h][e][l]
  vt[(vo + lane + 32) * LL + l] = (__bf16)v1;
}

// ---------------------------------------------------------------------------
// 6) Flash attention: 1 wave per (b,h,16 q-rows), 64-key chunks. Online
//    softmax; S and O accumulate in WMMA C layout; P goes through LDS for the
//    C->A relayout (fenced with s_wait_dscnt; same-wave LDS ops are in-order).
// ---------------------------------------------------------------------------
__global__ void attention_kernel(const __bf16* __restrict__ qb,
                                 const __bf16* __restrict__ kb,
                                 const __bf16* __restrict__ vt,
                                 const int* __restrict__ mask,
                                 __bf16* __restrict__ ao) {
  __shared__ __bf16 plds[8 * 16 * 64];            // 16 KB: 16x64 P tile per wave
  int wv = threadIdx.x >> 5, lane = threadIdx.x & 31;
  int gw = blockIdx.x * 8 + wv;                   // over B*H*(L/16) = 2048
  int qt = gw % (LL / 16), h = (gw / (LL / 16)) % HH, b = gw / ((LL / 16) * HH);
  int ml = lane & 15, kh = lane >> 4;
  size_t bh = (size_t)(b * HH + h);

  // q fragments (K = 0..31, 32..63)
  const __bf16* qrow = qb + (bh * LL + qt * 16 + ml) * EE;
  v16bf qa[2];
#pragma unroll
  for (int c = 0; c < 2; ++c) qa[c] = load_a_frag(qrow + c * 32 + kh * 16);

  float rm[8], rl[8];
#pragma unroll
  for (int r = 0; r < 8; ++r) { rm[r] = -1e30f; rl[r] = 0.f; }
  v8f oacc[4] = {};
  __bf16* pl = plds + wv * (16 * 64);

  for (int s0 = 0; s0 < LL; s0 += 64) {
    // --- S = q @ k^T (16x64), 8 WMMAs ---
    v8f st[4];
    float bias[4];
#pragma unroll
    for (int nt = 0; nt < 4; ++nt) {
      v8f a = {};
      int n = s0 + nt * 16 + ml;
#pragma unroll
      for (int c = 0; c < 2; ++c) {
        const __bf16* kp = kb + (bh * LL + n) * EE + c * 32 + kh * 16;
        a = wmma_bf16(qa[c], *(const v16bf*)kp, a);
      }
      st[nt] = a;
      bias[nt] = (mask[b * LL + n] == 1) ? 0.f : -1e9f;
    }

    // --- online softmax (row stats per (vgpr r, lane-half)) ---
    float scale[8];
#pragma unroll
    for (int r = 0; r < 8; ++r) {
      float a0 = st[0][r] + bias[0], a1 = st[1][r] + bias[1];
      float a2 = st[2][r] + bias[2], a3 = st[3][r] + bias[3];
      st[0][r] = a0; st[1][r] = a1; st[2][r] = a2; st[3][r] = a3;
      float mx = fmaxf(fmaxf(a0, a1), fmaxf(a2, a3));
#pragma unroll
      for (int off = 1; off < 16; off <<= 1) mx = fmaxf(mx, __shfl_xor(mx, off));
      float nm = fmaxf(rm[r], mx);
      scale[r] = __expf(rm[r] - nm);
      float p0 = __expf(a0 - nm), p1 = __expf(a1 - nm);
      float p2 = __expf(a2 - nm), p3 = __expf(a3 - nm);
      st[0][r] = p0; st[1][r] = p1; st[2][r] = p2; st[3][r] = p3;
      float rs = (p0 + p1) + (p2 + p3);
#pragma unroll
      for (int off = 1; off < 16; off <<= 1) rs += __shfl_xor(rs, off);
      rl[r] = rl[r] * scale[r] + rs;
      rm[r] = nm;
    }
#pragma unroll
    for (int nt = 0; nt < 4; ++nt)
#pragma unroll
      for (int r = 0; r < 8; ++r) oacc[nt][r] *= scale[r];

    // --- P: C layout -> LDS -> A layout (two 16x32 A-frags) ---
#pragma unroll
    for (int r = 0; r < 8; ++r) {
      int m = r + (kh ? 8 : 0);
#pragma unroll
      for (int nt = 0; nt < 4; ++nt)
        pl[m * 64 + nt * 16 + ml] = (__bf16)st[nt][r];
    }
    asm volatile("s_wait_dscnt 0" ::: "memory");
    v16bf pa[2];
#pragma unroll
    for (int c = 0; c < 2; ++c) pa[c] = load_a_frag(pl + ml * 64 + c * 32 + kh * 16);
    asm volatile("s_wait_dscnt 0" ::: "memory");

    // --- O += P @ V (K = 64 keys), 8 WMMAs ---
#pragma unroll
    for (int c = 0; c < 2; ++c)
#pragma unroll
      for (int nt = 0; nt < 4; ++nt) {
        const __bf16* vp =
            vt + (bh * EE + nt * 16 + ml) * LL + s0 + c * 32 + kh * 16;
        oacc[nt] = wmma_bf16(pa[c], *(const v16bf*)vp, oacc[nt]);
      }
  }

  // epilogue: o / l, write [b,l,h*64+e] as bf16
#pragma unroll
  for (int nt = 0; nt < 4; ++nt)
#pragma unroll
    for (int r = 0; r < 8; ++r) {
      int m = qt * 16 + r + (kh ? 8 : 0);
      int e = nt * 16 + ml;
      ao[((size_t)b * LL + m) * DD + h * EE + e] = (__bf16)(oacc[nt][r] / rl[r]);
    }
}

// ---------------------------------------------------------------------------
// 7) GLU: yg = a * gelu_exact(b)   (u:[ROWS,2*DFF] -> yg:[ROWS,DFF] bf16)
// ---------------------------------------------------------------------------
__global__ void glu_kernel(const float* __restrict__ u, __bf16* __restrict__ yg) {
  int idx = blockIdx.x * blockDim.x + threadIdx.x;   // ROWS*DFF
  if (idx >= ROWS * DFF) return;
  int j = idx % DFF, i = idx / DFF;
  float a  = u[(size_t)i * (2 * DFF) + j];
  float bv = u[(size_t)i * (2 * DFF) + DFF + j];
  float g  = 0.5f * bv * (1.0f + erff(bv * 0.70710678f));
  yg[idx] = (__bf16)(a * g);
}

// ---------------------------------------------------------------------------
// host orchestration
// ---------------------------------------------------------------------------
extern "C" void kernel_launch(void* const* d_in, const int* in_sizes, int n_in,
                              void* d_out, int out_size, void* d_ws, size_t ws_size,
                              hipStream_t stream) {
  const float* x        = (const float*)d_in[0];
  const float* cond     = (const float*)d_in[1];
  const int*   mask     = (const int*)  d_in[2];
  const float* wn_attn  = (const float*)d_in[3];
  const float* wqkv     = (const float*)d_in[4];
  const float* qk_scale = (const float*)d_in[5];
  const float* wout     = (const float*)d_in[6];
  const float* wn_ff    = (const float*)d_in[7];
  const float* wup      = (const float*)d_in[8];
  const float* wdown    = (const float*)d_in[9];
  const float* outscale = (const float*)d_in[10];

  char* ws = (char*)d_ws;
  size_t off = 0;
  auto carve = [&](size_t bytes) { size_t o = off; off = (off + bytes + 255) & ~(size_t)255; return o; };

  __bf16* wqkvT  = (__bf16*)(ws + carve((size_t)NLAY * DD * (3*DD)   * 2));
  __bf16* woutT  = (__bf16*)(ws + carve((size_t)NLAY * DD * DD       * 2));
  __bf16* wupT   = (__bf16*)(ws + carve((size_t)NLAY * DD * (2*DFF)  * 2));
  __bf16* wdownT = (__bf16*)(ws + carve((size_t)NLAY * DFF * DD      * 2));
  float*  g_attn = (float*) (ws + carve((size_t)NLAY * BB * DD * 4));
  float*  g_ff   = (float*) (ws + carve((size_t)NLAY * BB * DD * 4));
  float*  xbuf   = (float*) (ws + carve((size_t)ROWS * DD * 4));
  __bf16* ybf    = (__bf16*)(ws + carve((size_t)ROWS * DD * 2));
  float*  bigf   = (float*) (ws + carve((size_t)ROWS * (2*DFF) * 4)); // qkv / u share
  __bf16* qbf    = (__bf16*)(ws + carve((size_t)BB * HH * LL * EE * 2));
  __bf16* kbf    = (__bf16*)(ws + carve((size_t)BB * HH * LL * EE * 2));
  __bf16* vtbf   = (__bf16*)(ws + carve((size_t)BB * HH * EE * LL * 2));
  __bf16* aobf   = (__bf16*)(ws + carve((size_t)ROWS * DD * 2));
  __bf16* ygbf   = (__bf16*)(ws + carve((size_t)ROWS * DFF * 2));

  // ---- one-time passes (re-run every call: deterministic) ----
  for (int l = 0; l < NLAY; ++l) {
    w_to_bf16_t_kernel<<<(DD*3*DD)/256,  256, 0, stream>>>(wqkv  + (size_t)l*DD*3*DD,  wqkvT  + (size_t)l*3*DD*DD,  DD,  3*DD);
    w_to_bf16_t_kernel<<<(DD*DD)/256,    256, 0, stream>>>(wout  + (size_t)l*DD*DD,    woutT  + (size_t)l*DD*DD,    DD,  DD);
    w_to_bf16_t_kernel<<<(DD*2*DFF)/256, 256, 0, stream>>>(wup   + (size_t)l*DD*2*DFF, wupT   + (size_t)l*2*DFF*DD, DD,  2*DFF);
    w_to_bf16_t_kernel<<<(DFF*DD)/256,   256, 0, stream>>>(wdown + (size_t)l*DFF*DD,   wdownT + (size_t)l*DD*DFF,   DFF, DD);
  }
  cond_gain_kernel<<<(NLAY*BB*DD)/256, 256, 0, stream>>>(cond, wn_attn, g_attn);
  cond_gain_kernel<<<(NLAY*BB*DD)/256, 256, 0, stream>>>(cond, wn_ff,   g_ff);
  hipMemcpyAsync(xbuf, x, (size_t)ROWS * DD * 4, hipMemcpyDeviceToDevice, stream);

  // ---- transformer layers ----
  for (int l = 0; l < NLAY; ++l) {
    // attention branch
    rmsnorm_kernel<__bf16, true><<<ROWS/8, 256, 0, stream>>>(xbuf, g_attn + (size_t)l*BB*DD, ybf);
    gemm_bf16_wmma_kernel<<<(ROWS/32)*((3*DD)/64)/8, 256, 0, stream>>>(
        ybf, wqkvT + (size_t)l*3*DD*DD, nullptr, bigf, ROWS, 3*DD, DD);
    qk_prep_kernel<<<(BB*HH*LL)/8, 256, 0, stream>>>(bigf, qk_scale + (size_t)l*HH, qbf, kbf, vtbf);
    attention_kernel<<<(BB*HH*(LL/16))/8, 256, 0, stream>>>(qbf, kbf, vtbf, mask, aobf);
    gemm_bf16_wmma_kernel<<<(ROWS/32)*(DD/64)/8, 256, 0, stream>>>(
        aobf, woutT + (size_t)l*DD*DD, xbuf, xbuf, ROWS, DD, DD);

    // FF branch
    rmsnorm_kernel<__bf16, true><<<ROWS/8, 256, 0, stream>>>(xbuf, g_ff + (size_t)l*BB*DD, ybf);
    gemm_bf16_wmma_kernel<<<(ROWS/32)*((2*DFF)/64)/8, 256, 0, stream>>>(
        ybf, wupT + (size_t)l*2*DFF*DD, nullptr, bigf, ROWS, 2*DFF, DD);
    glu_kernel<<<(ROWS*DFF)/256, 256, 0, stream>>>(bigf, ygbf);
    gemm_bf16_wmma_kernel<<<(ROWS/32)*(DD/64)/8, 256, 0, stream>>>(
        ygbf, wdownT + (size_t)l*DD*DFF, xbuf, xbuf, ROWS, DD, DFF);
  }

  // final RMSNorm with shared out_scale, f32 output
  rmsnorm_kernel<float, false><<<ROWS/8, 256, 0, stream>>>(xbuf, outscale, (float*)d_out);
}
